// GCNEncoder3Layer_56616258895894
// MI455X (gfx1250) — matrix operand order repaired and checked by reading, not compiled, not run on hardware
//
#include <hip/hip_runtime.h>
#include <hip/hip_bf16.h>

typedef __bf16 bf16_t;
typedef __attribute__((ext_vector_type(16))) __bf16 v16bf;
typedef __attribute__((ext_vector_type(8)))  __bf16 v8bf;
typedef __attribute__((ext_vector_type(8)))  float  v8f;

static inline size_t align_up(size_t v, size_t a) { return (v + a - 1) & ~(a - 1); }

// ---------------- degree / normalization ----------------

__global__ void k_fill1(float* p, int n) {
    int i = blockIdx.x * blockDim.x + threadIdx.x;
    if (i < n) p[i] = 1.0f;                  // self-loop contributes 1 to degree
}

__global__ void k_deg(const int* __restrict__ ei, float* __restrict__ deg, int E) {
    int e = blockIdx.x * blockDim.x + threadIdx.x;
    if (e < E) atomicAdd(&deg[ei[E + e]], 1.0f);   // col = targets
}

__global__ void k_rsqrt_inplace(float* p, int n) {
    int i = blockIdx.x * blockDim.x + threadIdx.x;
    if (i < n) p[i] = rsqrtf(p[i]);          // deg >= 1 always
}

__global__ void k_f2bf(const float* __restrict__ s, bf16_t* __restrict__ d, int n) {
    int i = blockIdx.x * blockDim.x + threadIdx.x;
    if (i < n) d[i] = (bf16_t)s[i];
}

// ---------------- weight packing for WMMA B operand ----------------
// B operand k order chosen here (wave32, 16x16x32; A uses the same): lane L holds column N = L%16,
// 16 consecutive K values: K = kb*32 + (L<16 ? 0 : 16) + j, j=0..15.
// Pack so each lane's 16 bf16 values are contiguous (one 32B load).
template<int K, int N>
__global__ void k_packW(const float* __restrict__ W, bf16_t* __restrict__ Wp) {
    int idx = blockIdx.x * blockDim.x + threadIdx.x;
    if (idx >= K * N) return;
    constexpr int TN = N / 16;
    int j    = idx & 15;
    int lane = (idx >> 4) & 31;
    int rest = idx >> 9;
    int n_tile = rest % TN;
    int kb     = rest / TN;
    int half = lane >> 4, lrow = lane & 15;
    int k = kb * 32 + half * 16 + j;
    int n = n_tile * 16 + lrow;
    Wp[idx] = (bf16_t)W[k * N + n];
}

// ---------------- bf16 WMMA GEMM: out[nrows x N] = A[nrows x K] @ W ----------------
// Block = 256 threads (8 waves), covers a 64-row stripe (4 m-tiles).
// A stripe staged once in LDS (64 x K bf16 = 16 KB); B panel held in registers
// per wave and reused across its m-tiles.
template<int K, int N>
__global__ __launch_bounds__(256) void k_gemm_bf16(const bf16_t* __restrict__ A,
                                                   const bf16_t* __restrict__ Wp,
                                                   bf16_t* __restrict__ out,
                                                   int nrows) {
    constexpr int TN   = N / 16;       // n tiles (8 or 4)
    constexpr int WG   = 8 / TN;       // wave groups along M (1 or 2)
    constexpr int MTPW = 4 / WG;       // m-tiles per wave (4 or 2)
    constexpr int KB   = K / 32;       // wmma steps along K

    __shared__ __align__(16) bf16_t As[64 * K];

    const int tid  = threadIdx.x;
    const int lane = tid & 31;
    const int wave = tid >> 5;
    const int n_tile = wave % TN;
    const int half = lane >> 4, lrow = lane & 15;
    const int blockRow0 = blockIdx.x * 64;

    // ---- preload this wave's B panel (reused across all its m-tiles) ----
    v16bf breg[KB];
#pragma unroll
    for (int kb = 0; kb < KB; ++kb)
        breg[kb] = *(const v16bf*)(Wp + ((size_t)(kb * TN + n_tile) * 32 + lane) * 16);

    // ---- cooperative stage of the 64 x K A stripe into LDS ----
#pragma unroll
    for (int it = 0; it < (64 * K / 8) / 256; ++it) {
        int chunk = it * 256 + tid;          // 8-element (16B) chunk id
        int r   = chunk / (K / 8);
        int off = (chunk % (K / 8)) * 8;
        int grow = blockRow0 + r;
        if (grow < nrows)
            *(v8bf*)&As[r * K + off] = *(const v8bf*)&A[(size_t)grow * K + off];
    }
    __syncthreads();

    // ---- compute: each wave sweeps its m-tiles with B resident ----
#pragma unroll
    for (int i = 0; i < MTPW; ++i) {
        int mt   = (wave / TN) * MTPW + i;
        int row0 = blockRow0 + mt * 16;
        if (row0 >= nrows) continue;         // uniform per wave; tiles are whole

        // A layout (matches packed B): lanes 0-15 -> row M=lane, K in {0..15};
        // lanes 16-31 -> row M=lane-16, K in {16..31}.
        const bf16_t* arow = As + (mt * 16 + lrow) * K + half * 16;
        v8f acc = {};
#pragma unroll
        for (int kb = 0; kb < KB; ++kb) {
            v8bf alo = *(const v8bf*)(arow + kb * 32);        // ds_load_b128
            v8bf ahi = *(const v8bf*)(arow + kb * 32 + 8);
            v16bf a;
#pragma unroll
            for (int j = 0; j < 8; ++j) { a[j] = alo[j]; a[j + 8] = ahi[j]; }
            acc = __builtin_amdgcn_wmma_f32_16x16x32_bf16(
                /*neg_a=*/false, a, /*neg_b=*/false, breg[kb],
                /*c_mod=*/(short)0, acc, /*reuse_a=*/false, /*reuse_b=*/false);
        }
        // D layout: VGPR v -> row M = v + half*8, col = lrow
        bf16_t* orow = out + (size_t)row0 * N + n_tile * 16 + lrow;
#pragma unroll
        for (int v = 0; v < 8; ++v)
            orow[(size_t)(v + half * 8) * N] = (bf16_t)acc[v];
    }
}

// ---------------- aggregation ----------------

// self-loop + zero-init: acc[i][c] = tmp[i][c] * dinv[i]^2
template<int C>
__global__ void k_agg_init(const bf16_t* __restrict__ tmp, const float* __restrict__ dinv,
                           float* __restrict__ acc, int n) {
    int idx = blockIdx.x * blockDim.x + threadIdx.x;
    if (idx >= n * C) return;
    int i = idx / C;
    float dv = dinv[i];
    acc[idx] = (float)tmp[idx] * dv * dv;
}

// edge scatter-add: acc[col][c] += tmp[row][c] * dinv[row]*dinv[col]
template<int C>
__global__ void k_agg_edge(const int* __restrict__ ei, const bf16_t* __restrict__ tmp,
                           const float* __restrict__ dinv, float* __restrict__ acc, int E) {
    constexpr int CH = C / 8;
    int t = blockIdx.x * blockDim.x + threadIdx.x;
    if (t >= E * CH) return;
    int e  = t / CH;
    int ch = (t % CH) * 8;
    int r = ei[e];
    int c = ei[E + e];
    float nrm = dinv[r] * dinv[c];
    v8bf m = *(const v8bf*)(tmp + (size_t)r * C + ch);   // 16B L2-resident gather
    float* dst = acc + (size_t)c * C + ch;
#pragma unroll
    for (int j = 0; j < 8; ++j)
        atomicAdd(dst + j, (float)m[j] * nrm);           // global_atomic_add_f32
}

template<int C>
__global__ void k_bias_relu_bf16(const float* __restrict__ acc, const float* __restrict__ b,
                                 bf16_t* __restrict__ h, int n) {
    int idx = blockIdx.x * blockDim.x + threadIdx.x;
    if (idx >= n * C) return;
    float v = acc[idx] + b[idx % C];
    h[idx] = (bf16_t)(v > 0.0f ? v : 0.0f);
}

template<int C>
__global__ void k_bias_out_f32(const float* __restrict__ acc, const float* __restrict__ b,
                               float* __restrict__ out, int n) {
    int idx = blockIdx.x * blockDim.x + threadIdx.x;
    if (idx >= n * C) return;
    out[idx] = acc[idx] + b[idx % C];
}

// ---------------- launch ----------------

extern "C" void kernel_launch(void* const* d_in, const int* in_sizes, int n_in,
                              void* d_out, int out_size, void* d_ws, size_t ws_size,
                              hipStream_t stream) {
    const float* x  = (const float*)d_in[0];
    const int*   ei = (const int*)d_in[1];
    const float* W1 = (const float*)d_in[2];
    const float* b1 = (const float*)d_in[3];
    const float* W2 = (const float*)d_in[4];
    const float* b2 = (const float*)d_in[5];
    const float* W3 = (const float*)d_in[6];
    const float* b3 = (const float*)d_in[7];

    constexpr int IN_C = 128, HID = 128, OUT = 64;
    const int n = in_sizes[0] / IN_C;   // 100000
    const int E = in_sizes[1] / 2;      // 1600000

    char* ws = (char*)d_ws;
    size_t off = 0;
    float*  dinv = (float*)(ws + off);  off = align_up(off + (size_t)n * 4, 256);
    bf16_t* hbf  = (bf16_t*)(ws + off); off = align_up(off + (size_t)n * HID * 2, 256);
    bf16_t* tbf  = (bf16_t*)(ws + off); off = align_up(off + (size_t)n * HID * 2, 256);
    float*  acc  = (float*)(ws + off);  off = align_up(off + (size_t)n * HID * 4, 256);
    bf16_t* Wp1  = (bf16_t*)(ws + off); off = align_up(off + (size_t)IN_C * HID * 2, 256);
    bf16_t* Wp2  = (bf16_t*)(ws + off); off = align_up(off + (size_t)HID * HID * 2, 256);
    bf16_t* Wp3  = (bf16_t*)(ws + off); off = align_up(off + (size_t)HID * OUT * 2, 256);
    (void)ws_size;

    const int T = 256;
    auto blk = [&](long long cnt) { return (int)((cnt + T - 1) / T); };

    // --- normalization ---
    k_fill1<<<blk(n), T, 0, stream>>>(dinv, n);
    k_deg<<<blk(E), T, 0, stream>>>(ei, dinv, E);
    k_rsqrt_inplace<<<blk(n), T, 0, stream>>>(dinv, n);

    // --- weight packing + input conversion ---
    k_packW<IN_C, HID><<<blk(IN_C * HID), T, 0, stream>>>(W1, Wp1);
    k_packW<HID, HID><<<blk(HID * HID), T, 0, stream>>>(W2, Wp2);
    k_packW<HID, OUT><<<blk(HID * OUT), T, 0, stream>>>(W3, Wp3);
    k_f2bf<<<blk((long long)n * IN_C), T, 0, stream>>>(x, hbf, n * IN_C);

    const int gemm_blocks = (n + 63) / 64;   // 64-row stripe per block

    // --- layer 1 ---
    k_gemm_bf16<IN_C, HID><<<gemm_blocks, T, 0, stream>>>(hbf, Wp1, tbf, n);
    k_agg_init<HID><<<blk((long long)n * HID), T, 0, stream>>>(tbf, dinv, acc, n);
    k_agg_edge<HID><<<blk((long long)E * (HID / 8)), T, 0, stream>>>(ei, tbf, dinv, acc, E);
    k_bias_relu_bf16<HID><<<blk((long long)n * HID), T, 0, stream>>>(acc, b1, hbf, n);

    // --- layer 2 ---
    k_gemm_bf16<HID, HID><<<gemm_blocks, T, 0, stream>>>(hbf, Wp2, tbf, n);
    k_agg_init<HID><<<blk((long long)n * HID), T, 0, stream>>>(tbf, dinv, acc, n);
    k_agg_edge<HID><<<blk((long long)E * (HID / 8)), T, 0, stream>>>(ei, tbf, dinv, acc, E);
    k_bias_relu_bf16<HID><<<blk((long long)n * HID), T, 0, stream>>>(acc, b2, hbf, n);

    // --- layer 3 (no relu, f32 output) ---
    k_gemm_bf16<HID, OUT><<<gemm_blocks, T, 0, stream>>>(hbf, Wp3, tbf, n);
    k_agg_init<OUT><<<blk((long long)n * OUT), T, 0, stream>>>(tbf, dinv, acc, n);
    k_agg_edge<OUT><<<blk((long long)E * (OUT / 8)), T, 0, stream>>>(ei, tbf, dinv, acc, E);
    k_bias_out_f32<OUT><<<blk((long long)n * OUT), T, 0, stream>>>(acc, b3, (float*)d_out, n);
}
